// MinkowskiEncoder_31653908972122
// MI455X (gfx1250) — matrix operand order repaired and checked by reading, not compiled, not run on hardware
//
#include <hip/hip_runtime.h>

typedef _Float16 f16_t;
typedef __attribute__((ext_vector_type(16))) _Float16 v16h;
typedef __attribute__((ext_vector_type(8)))  _Float16 v8h;
typedef __attribute__((ext_vector_type(8)))  float    v8f;

#define BATCH 8

// ---------------------------------------------------------------------------
// Masked 3x3 stride-2 max pool ("SAME"): window rows 2y..2y+2, cols 2x..2x+2.
// Input NCHW f32; output NHWC f16 with channel stride ICP (>= C, zero-padded).
// maskin == nullptr -> stage-1 semantics (valid = x > 0 on the single channel).
// ---------------------------------------------------------------------------
__global__ void pool_kernel(const float* __restrict__ xin,
                            const float* __restrict__ maskin,
                            _Float16* __restrict__ xout,
                            float* __restrict__ maskout,
                            int C, int ICP, int Hin, int Win, int Hout, int Wout)
{
    long long idx  = (long long)blockIdx.x * blockDim.x + threadIdx.x;
    long long npix = (long long)BATCH * Hout * Wout;
    if (idx >= npix) return;
    int xo = (int)(idx % Wout);
    long long t = idx / Wout;
    int yo = (int)(t % Hout);
    int b  = (int)(t / Hout);

    unsigned vb = 0;
    #pragma unroll
    for (int dy = 0; dy < 3; ++dy)
        #pragma unroll
        for (int dx = 0; dx < 3; ++dx) {
            int yi = 2 * yo + dy, xi = 2 * xo + dx;
            if (yi < Hin && xi < Win) {
                float mv;
                if (maskin) mv = maskin[((long long)b * Hin + yi) * Win + xi];
                else        mv = (xin[((long long)b * Hin + yi) * Win + xi] > 0.f) ? 1.f : 0.f;
                if (mv > 0.f) vb |= 1u << (dy * 3 + dx);
            }
        }
    maskout[((long long)b * Hout + yo) * Wout + xo] = vb ? 1.f : 0.f;

    _Float16* orow = xout + ((long long)b * Hout + yo) * Wout * (long long)ICP
                          + (long long)xo * ICP;
    for (int c = 0; c < C; ++c) {
        float p = -1e30f;
        #pragma unroll
        for (int dy = 0; dy < 3; ++dy)
            #pragma unroll
            for (int dx = 0; dx < 3; ++dx)
                if (vb & (1u << (dy * 3 + dx))) {
                    int yi = 2 * yo + dy, xi = 2 * xo + dx;
                    float v = xin[(((long long)b * C + c) * Hin + yi) * Win + xi];
                    p = fmaxf(p, v);
                }
        orow[c] = (f16_t)(vb ? p : 0.f);
    }
    for (int c = C; c < ICP; ++c) orow[c] = (f16_t)0.f;   // K-dim zero padding
}

// ---------------------------------------------------------------------------
// Weight repack to f16 in B-fragment order.
//   dst[(((g*TAPS + tap)*NCH + kc)*16 + n)*32 + k], zero for icp >= IC.
// Source index w[(oc*IC + icp)*TAPS + tap] covers both normal convs
// (IC=in-ch, TAPS=ks*ks) and stage-1 IC=1 convs (IC=25 "channels", TAPS=1).
// ---------------------------------------------------------------------------
__global__ void pack_w_kernel(const float* __restrict__ w,
                              _Float16* __restrict__ dst,
                              int OC, int IC, int ICP, int TAPS)
{
    long long i = (long long)blockIdx.x * blockDim.x + threadIdx.x;
    long long total = (long long)OC * TAPS * ICP;
    if (i >= total) return;
    int icp = (int)(i % ICP);
    long long t = i / ICP;
    int tap = (int)(t % TAPS);
    int oc  = (int)(t / TAPS);
    int g = oc >> 4, n = oc & 15;
    int kc = icp >> 5, k = icp & 31;
    int NCH = ICP >> 5;
    float v = (icp < IC) ? w[((long long)oc * IC + icp) * TAPS + tap] : 0.f;
    dst[((((long long)g * TAPS + tap) * NCH + kc) * 16 + n) * 32 + k] = (f16_t)v;
}

// ---------------------------------------------------------------------------
// Inner MACs: per-tap GEMM over channel chunks of 32; f16 NHWC activations.
// A fragment = two 16B packed-half loads; B fragment = one 32B v16h load.
// CHECK=false -> interior tile, no per-lane bounds checks, immediate offsets.
// ---------------------------------------------------------------------------
template<int ICP, int KS, bool CHECK>
__device__ __forceinline__ void conv_inner(const _Float16* __restrict__ x,
                                           const _Float16* __restrict__ wg,
                                           int b, int y, int x0, int am, int hi,
                                           int H, int W, v8f& acc)
{
    constexpr int NCH = ICP / 32;
    constexpr int PAD = (KS - 1) / 2;

    for (int ky = 0; ky < KS; ++ky) {
        const int iy = y + ky - PAD;
        if (iy < 0 || iy >= H) continue;
        const _Float16* xb = x
            + ((long long)(b * H + iy) * W + (x0 + am - PAD)) * (long long)ICP
            + hi * 16;
        const _Float16* wrow = wg + (long long)ky * KS * NCH * 512;
        #pragma unroll
        for (int kx = 0; kx < KS; ++kx) {
            bool inb = true;
            if (CHECK) {
                int ix = x0 + am + kx - PAD;
                inb = (ix >= 0) && (ix < W);
            }
            #pragma unroll
            for (int kc = 0; kc < NCH; ++kc) {
                v16h afrag;
                if (!CHECK || inb) {
                    const _Float16* p0 = xb + kx * ICP + kc * 32;
                    v8h a0 = *(const v8h*)(p0);        // K = kc*32 + hi*16 ..
                    v8h a1 = *(const v8h*)(p0 + 8);    // K = kc*32 + hi*16 + 8 ..
                    #pragma unroll
                    for (int e = 0; e < 8; ++e) {
                        afrag[e]     = a0[e];
                        afrag[8 + e] = a1[e];
                    }
                } else {
                    #pragma unroll
                    for (int e = 0; e < 16; ++e) afrag[e] = (f16_t)0.f;
                }
                v16h bfrag = *(const v16h*)(wrow + kx * NCH * 512 + kc * 512);

                acc = __builtin_amdgcn_wmma_f32_16x16x32_f16(
                          false, afrag, false, bfrag, (short)0, acc, false, false);
            }
        }
    }
}

// Shared masked-store epilogue (f32 output): uniform fast path with vector
// mask loads and immediate-offset stores; guarded path only for ragged tiles.
template<int OC>
__device__ __forceinline__ void conv_epilogue(const v8f& acc,
                                              const float* __restrict__ maskf,
                                              float* __restrict__ out,
                                              int b, int y, int x0, int am,
                                              int hi, int ocb, int H, int W,
                                              int accumulate)
{
    const long long rowp = (long long)(b * H + y) * W + x0 + hi * 8;
    float* obase = out + rowp * OC + ocb + am;
    const float* mrow = maskf + rowp;

    if (x0 + 15 < W) {
        float4 m0 = *(const float4*)(mrow);
        float4 m1 = *(const float4*)(mrow + 4);
        float mv[8] = { m0.x, m0.y, m0.z, m0.w, m1.x, m1.y, m1.z, m1.w };
        if (accumulate) {
            #pragma unroll
            for (int r = 0; r < 8; ++r) obase[r * OC] += acc[r] * mv[r];
        } else {
            #pragma unroll
            for (int r = 0; r < 8; ++r) obase[r * OC] = acc[r] * mv[r];
        }
    } else {
        #pragma unroll
        for (int r = 0; r < 8; ++r) {
            int xx = x0 + hi * 8 + r;
            if (xx < W) {
                float v = acc[r] * mrow[r];
                if (accumulate) obase[r * OC] += v;
                else            obase[r * OC]  = v;
            }
        }
    }
}

// ---------------------------------------------------------------------------
// NHWC implicit-GEMM conv via V_WMMA_F32_16X16X32_F16.
// Grid: x = ceil(tiles_x * OC/16 / 8 waves), y = H, z = batch.
// ---------------------------------------------------------------------------
template<int ICP, int OC, int KS>
__global__ __launch_bounds__(256)
void conv_nhwc_wmma_kernel(const _Float16* __restrict__ x,
                           const _Float16* __restrict__ wp,
                           const float* __restrict__ maskf,
                           float* __restrict__ out,
                           int H, int W, int accumulate)
{
    constexpr int TAPS = KS * KS;
    constexpr int NCH  = ICP / 32;
    constexpr int PAD  = (KS - 1) / 2;
    constexpr int OCG  = OC / 16;          // power of two

    const int lane = threadIdx.x & 31;
    const int wave = threadIdx.x >> 5;
    const int tiles_x = (W + 15) >> 4;

    const int q = blockIdx.x * (blockDim.x >> 5) + wave;
    if (q >= tiles_x * OCG) return;
    const int g  = q & (OCG - 1);
    const int tx = q / OCG;                // constexpr power-of-two shift
    const int y  = blockIdx.y;
    const int b  = blockIdx.z;
    const int x0  = tx << 4;
    const int ocb = g << 4;

    const int hi = (lane >> 4) & 1;
    const int am = lane & 15;

    const _Float16* wg = wp + (long long)g * TAPS * NCH * 512 + am * 32 + hi * 16;

    v8f acc = {};
    const bool interior = (x0 > 0) && (x0 + 15 + PAD < W);
    if (interior)
        conv_inner<ICP, KS, false>(x, wg, b, y, x0, am, hi, H, W, acc);
    else
        conv_inner<ICP, KS, true >(x, wg, b, y, x0, am, hi, H, W, acc);

    conv_epilogue<OC>(acc, maskf, out, b, y, x0, am, hi, ocb, H, W, accumulate);
}

// ---------------------------------------------------------------------------
// Stage-1 (single input channel, f16) conv: K = 25 taps in one WMMA chunk.
// ---------------------------------------------------------------------------
template<int OC>
__global__ __launch_bounds__(256)
void conv_ic1_wmma_kernel(const _Float16* __restrict__ x,
                          const _Float16* __restrict__ wp,
                          const float* __restrict__ maskf,
                          float* __restrict__ out,
                          int H, int W)
{
    constexpr int KS = 5, KK = 25, PAD = 2;
    constexpr int OCG = OC / 16;

    const int lane = threadIdx.x & 31;
    const int wave = threadIdx.x >> 5;
    const int tiles_x = (W + 15) >> 4;

    const int q = blockIdx.x * (blockDim.x >> 5) + wave;
    if (q >= tiles_x * OCG) return;
    const int g  = q & (OCG - 1);
    const int tx = q / OCG;
    const int y  = blockIdx.y;
    const int b  = blockIdx.z;
    const int x0  = tx << 4;
    const int ocb = g << 4;

    const int hi = (lane >> 4) & 1;
    const int am = lane & 15;

    v16h afrag;
    #pragma unroll
    for (int e = 0; e < 16; ++e) {
        int kg = e + (hi ? 16 : 0);
        f16_t av = (f16_t)0.f;
        if (kg < KK) {
            int ky = kg / KS, kx = kg % KS;
            int iy = y + ky - PAD, ix = x0 + am + kx - PAD;
            if (iy >= 0 && iy < H && ix >= 0 && ix < W)
                av = x[(long long)(b * H + iy) * W + ix];
        }
        afrag[e] = av;
    }
    v16h bfrag = *(const v16h*)(wp + (g * 16 + am) * 32 + hi * 16);

    v8f acc = {};
    acc = __builtin_amdgcn_wmma_f32_16x16x32_f16(
              false, afrag, false, bfrag, (short)0, acc, false, false);

    conv_epilogue<OC>(acc, maskf, out, b, y, x0, am, hi, ocb, H, W, 0);
}

// ---------------------------------------------------------------------------
// Utility kernels
// ---------------------------------------------------------------------------
__global__ void zero_kernel(float* __restrict__ p, long long n)
{
    long long i = (long long)blockIdx.x * blockDim.x + threadIdx.x;
    if (i < n) p[i] = 0.f;
}

__global__ void count_kernel(const float* __restrict__ m, long long n,
                             float* __restrict__ cnt)
{
    float s = 0.f;
    for (long long i = (long long)blockIdx.x * blockDim.x + threadIdx.x; i < n;
         i += (long long)gridDim.x * blockDim.x)
        s += (m[i] > 0.f) ? 1.f : 0.f;
    __shared__ float ls[256];
    ls[threadIdx.x] = s;
    __syncthreads();
    for (int st = 128; st > 0; st >>= 1) {
        if ((int)threadIdx.x < st) ls[threadIdx.x] += ls[threadIdx.x + st];
        __syncthreads();
    }
    if (threadIdx.x == 0) atomicAdd(cnt, ls[0]);
}

// Per-channel sum / sum^2 over f32 NHWC tensor. blockIdx.y = channel.
__global__ void stats_kernel(const float* __restrict__ x,
                             float* __restrict__ stats,
                             int C, int H, int W)
{
    int c = blockIdx.y;
    long long n = (long long)BATCH * H * W;
    float s = 0.f, s2 = 0.f;
    for (long long p = (long long)blockIdx.x * blockDim.x + threadIdx.x; p < n;
         p += (long long)gridDim.x * blockDim.x) {
        float v = x[p * C + c];
        s += v; s2 += v * v;
    }
    __shared__ float ls[256], ls2[256];
    ls[threadIdx.x] = s; ls2[threadIdx.x] = s2;
    __syncthreads();
    for (int st = 128; st > 0; st >>= 1) {
        if ((int)threadIdx.x < st) {
            ls[threadIdx.x]  += ls[threadIdx.x + st];
            ls2[threadIdx.x] += ls2[threadIdx.x + st];
        }
        __syncthreads();
    }
    if (threadIdx.x == 0) {
        atomicAdd(&stats[2 * c],     ls[0]);
        atomicAdd(&stats[2 * c + 1], ls2[0]);
    }
}

// Masked BN (+ReLU). Input f32 NHWC.
// mode 0: write f16 NHWC (next conv input); mode 1: write f32 NCHW (d_out).
__global__ void bn_apply_kernel(const float* __restrict__ x,
                                const float* __restrict__ maskf,
                                const float* __restrict__ stats,
                                const float* __restrict__ cnt,
                                const float* __restrict__ gamma,
                                const float* __restrict__ beta,
                                void* __restrict__ out,
                                int C, int H, int W, int relu, int mode)
{
    long long hw = (long long)H * W;
    long long n  = (long long)BATCH * hw * C;
    long long i  = (long long)blockIdx.x * blockDim.x + threadIdx.x;
    if (i >= n) return;
    int c = (int)(i % C);
    long long p = i / C;               // global pixel = b*hw + r
    float N   = cnt[0];
    float mu  = stats[2 * c] / N;
    float var = stats[2 * c + 1] / N - mu * mu;
    var = var < 0.f ? 0.f : var;
    float inv = rsqrtf(var + 1e-5f);
    float mf  = maskf[p];
    float v   = ((x[i] - mu) * inv * gamma[c] + beta[c]) * mf;
    if (relu) v = v > 0.f ? v : 0.f;
    if (mode) {
        long long b = p / hw, r = p % hw;
        ((float*)out)[(b * C + c) * hw + r] = v;
    } else {
        ((_Float16*)out)[i] = (f16_t)v;
    }
}

// ---------------------------------------------------------------------------
// Host-side stage driver.
// Param dict order per stage (14 entries):
//  0:l1_w 1:l2_w1 2:l2_g1 3:l2_b1 4:l2_w2 5:l3_w1 6:l3_g1 7:l3_b1
//  8:l3_w2 9:l3_g2 10:l3_b2 11:l3_w3 12:f_g 13:f_b
// ---------------------------------------------------------------------------
static inline long long llmin2(long long a, long long b) { return a < b ? a : b; }

template<int CIN, int CINP, int C, int KS, bool FIRST>
static void run_stage(const float* xin_nchw, const float* maskin,
                      int Hin, int Win, int H, int W,
                      void* const* d_in, int pbase,
                      _Float16* XPH, float* MASK, float* ACC, float* SF,
                      _Float16* S1H, _Float16* S2H,
                      float* STATS, float* CNT, _Float16* WPK,
                      float* outp, hipStream_t stream)
{
    constexpr int T = KS * KS;
    const float* w_l1   = (const float*)d_in[pbase + 0];
    const float* w_l2w1 = (const float*)d_in[pbase + 1];
    const float* l2_g1  = (const float*)d_in[pbase + 2];
    const float* l2_b1  = (const float*)d_in[pbase + 3];
    const float* w_l2w2 = (const float*)d_in[pbase + 4];
    const float* w_l3w1 = (const float*)d_in[pbase + 5];
    const float* l3_g1  = (const float*)d_in[pbase + 6];
    const float* l3_b1  = (const float*)d_in[pbase + 7];
    const float* w_l3w2 = (const float*)d_in[pbase + 8];
    const float* l3_g2  = (const float*)d_in[pbase + 9];
    const float* l3_b2  = (const float*)d_in[pbase + 10];
    const float* w_l3w3 = (const float*)d_in[pbase + 11];
    const float* f_g    = (const float*)d_in[pbase + 12];
    const float* f_b    = (const float*)d_in[pbase + 13];

    const int TB = 256;
    long long npix = (long long)BATCH * H * W;

    // ---- pack all six weight tensors (f16, B-fragment order) ----
    size_t po = 0;
    auto pack = [&](const float* w, int OC, int IC, int ICP, int TAPS) {
        _Float16* dst = WPK + po;
        long long tot = (long long)OC * TAPS * ICP;
        po += (size_t)tot;
        pack_w_kernel<<<(unsigned)((tot + TB - 1) / TB), TB, 0, stream>>>(
            w, dst, OC, IC, ICP, TAPS);
        return (const _Float16*)dst;
    };
    const _Float16 *pk_l1, *pk_l2w1, *pk_l3w1;
    if (FIRST) {
        pk_l1   = pack(w_l1,   C,     25, 32, 1);   // kg-packed, 1 chunk
        pk_l2w1 = pack(w_l2w1, 2 * C, 25, 32, 1);
        pk_l3w1 = pack(w_l3w1, 2 * C, 25, 32, 1);
    } else {
        pk_l1   = pack(w_l1,   C,     CIN, CINP, T);
        pk_l2w1 = pack(w_l2w1, 2 * C, CIN, CINP, T);
        pk_l3w1 = pack(w_l3w1, 2 * C, CIN, CINP, T);
    }
    const _Float16* pk_l2w2 = pack(w_l2w2, C,     2 * C, 2 * C, T);
    const _Float16* pk_l3w2 = pack(w_l3w2, 2 * C, 2 * C, 2 * C, T);
    const _Float16* pk_l3w3 = pack(w_l3w3, C,     2 * C, 2 * C, T);

    // ---- pool (NCHW f32 in -> NHWC f16 padded out) + mask + valid count ----
    constexpr int XICP = FIRST ? 1 : CINP;
    pool_kernel<<<(unsigned)((npix + TB - 1) / TB), TB, 0, stream>>>(
        xin_nchw, maskin, XPH, MASK, FIRST ? 1 : CIN, XICP, Hin, Win, H, W);
    zero_kernel<<<1, 64, 0, stream>>>(CNT, 1);
    count_kernel<<<(unsigned)llmin2((npix + TB - 1) / TB, 1024), TB, 0, stream>>>(
        MASK, npix, CNT);

    auto cgrid = [&](int OC) {
        long long nt = (long long)((W + 15) / 16) * (OC / 16);
        return dim3((unsigned)((nt + 7) / 8), (unsigned)H, BATCH);
    };
    auto run_bn = [&](const float* buf, int CC, const float* g, const float* bb,
                      void* dst, int mode) {
        zero_kernel<<<(unsigned)((2 * CC + TB - 1) / TB), TB, 0, stream>>>(STATS, 2 * CC);
        dim3 sg((unsigned)llmin2((npix + TB - 1) / TB, 256), (unsigned)CC);
        stats_kernel<<<sg, TB, 0, stream>>>(buf, STATS, CC, H, W);
        long long n = npix * CC;
        bn_apply_kernel<<<(unsigned)((n + TB - 1) / TB), TB, 0, stream>>>(
            buf, MASK, STATS, CNT, g, bb, dst, CC, H, W, 1, mode);
    };

    // ---- layer1: x1 = conv(x) -> ACC (overwrite) ----
    if (FIRST)
        conv_ic1_wmma_kernel<C><<<cgrid(C), TB, 0, stream>>>(
            XPH, pk_l1, MASK, ACC, H, W);
    else
        conv_nhwc_wmma_kernel<CINP, C, KS><<<cgrid(C), TB, 0, stream>>>(
            XPH, pk_l1, MASK, ACC, H, W, 0);

    // ---- layer2: relu(bn(conv(x))) -> conv -> += ACC ----
    if (FIRST)
        conv_ic1_wmma_kernel<2 * C><<<cgrid(2 * C), TB, 0, stream>>>(
            XPH, pk_l2w1, MASK, SF, H, W);
    else
        conv_nhwc_wmma_kernel<CINP, 2 * C, KS><<<cgrid(2 * C), TB, 0, stream>>>(
            XPH, pk_l2w1, MASK, SF, H, W, 0);
    run_bn(SF, 2 * C, l2_g1, l2_b1, S1H, 0);
    conv_nhwc_wmma_kernel<2 * C, C, KS><<<cgrid(C), TB, 0, stream>>>(
        S1H, pk_l2w2, MASK, ACC, H, W, 1);

    // ---- layer3: BN-ReLU-conv x2, then += ACC ----
    if (FIRST)
        conv_ic1_wmma_kernel<2 * C><<<cgrid(2 * C), TB, 0, stream>>>(
            XPH, pk_l3w1, MASK, SF, H, W);
    else
        conv_nhwc_wmma_kernel<CINP, 2 * C, KS><<<cgrid(2 * C), TB, 0, stream>>>(
            XPH, pk_l3w1, MASK, SF, H, W, 0);
    run_bn(SF, 2 * C, l3_g1, l3_b1, S1H, 0);
    conv_nhwc_wmma_kernel<2 * C, 2 * C, KS><<<cgrid(2 * C), TB, 0, stream>>>(
        S1H, pk_l3w2, MASK, SF, H, W, 0);
    run_bn(SF, 2 * C, l3_g2, l3_b2, S2H, 0);
    conv_nhwc_wmma_kernel<2 * C, C, KS><<<cgrid(C), TB, 0, stream>>>(
        S2H, pk_l3w3, MASK, ACC, H, W, 1);

    // ---- layer_final: BN-ReLU over sum -> d_out segment (f32 NCHW) ----
    run_bn(ACC, C, f_g, f_b, outp, 1);
}

// ---------------------------------------------------------------------------
// Entry point
// ---------------------------------------------------------------------------
extern "C" void kernel_launch(void* const* d_in, const int* in_sizes, int n_in,
                              void* d_out, int out_size, void* d_ws, size_t ws_size,
                              hipStream_t stream)
{
    (void)in_sizes; (void)n_in; (void)out_size; (void)ws_size;

    const float* d = (const float*)d_in[0];
    float* out = (float*)d_out;
    float* ws  = (float*)d_ws;

    // workspace arenas (float-sized offsets), sized for the worst stage
    const size_t PL1 = (size_t)BATCH * 176 * 608;        // 856064: max pooled plane
    size_t o = 0;
    _Float16* XPH = (_Float16*)(ws + o); o += 3424256;   // max pixels*ICP halves (stage2)
    float* MA    = ws + o; o += PL1;                     // mask ping
    float* MB    = ws + o; o += PL1;                     // mask pong
    float* ACC   = ws + o; o += 13697024;                // max OC*plane f32 (stage1)
    float* SF    = ws + o; o += 27394048;                // 2OC*plane f32 conv scratch
    _Float16* S1H = (_Float16*)(ws + o); o += 13697024;  // 2OC*plane f16
    _Float16* S2H = (_Float16*)(ws + o); o += 13697024;  // 2OC*plane f16
    float* STATS = ws + o; o += 512;                     // 2*Cmax(256)
    float* CNT   = ws + o; o += 16;
    _Float16* WPK = (_Float16*)(ws + o); o += 1048576;   // 2M f16 packed weights

    float* out1 = out;
    float* out2 = out1 + (size_t)BATCH * 16  * 176 * 608;
    float* out3 = out2 + (size_t)BATCH * 32  * 88  * 304;
    float* out4 = out3 + (size_t)BATCH * 64  * 44  * 152;

    // stage s params start at 1 + 14*s
    run_stage<1,   32,  16, 5, true >(d,    nullptr, 352, 1216, 176, 608, d_in, 1,
                                      XPH, MA, ACC, SF, S1H, S2H, STATS, CNT, WPK, out1, stream);
    run_stage<16,  32,  32, 5, false>(out1, MA,      176, 608,  88,  304, d_in, 1 + 14,
                                      XPH, MB, ACC, SF, S1H, S2H, STATS, CNT, WPK, out2, stream);
    run_stage<32,  32,  64, 3, false>(out2, MB,      88,  304,  44,  152, d_in, 1 + 28,
                                      XPH, MA, ACC, SF, S1H, S2H, STATS, CNT, WPK, out3, stream);
    run_stage<64,  64, 128, 3, false>(out3, MA,      44,  152,  22,  76,  d_in, 1 + 42,
                                      XPH, MB, ACC, SF, S1H, S2H, STATS, CNT, WPK, out4, stream);
}
